// GraphEncoder_4389456577106
// MI455X (gfx1250) — matrix operand order repaired and checked by reading, not compiled, not run on hardware
//
#include <hip/hip_runtime.h>
#include <hip/hip_bf16.h>

typedef __attribute__((ext_vector_type(16))) _Float16 v16h;
typedef __attribute__((ext_vector_type(8)))  float    v8f;

#define N_NODES 100000
#define E_EDGES 1600000
#define IN_DIM  128
#define HEADS   4
#define DDIM    32
#define NEG_SLOPE 0.2f

#define ROWS_PER_BLOCK 64
#define N_BLOCKS ((N_NODES + ROWS_PER_BLOCK - 1) / ROWS_PER_BLOCK)   // 1563

__device__ __forceinline__ unsigned enc_f32(float f) {
    unsigned u = __float_as_uint(f);
    return (u & 0x80000000u) ? ~u : (u | 0x80000000u);
}
__device__ __forceinline__ float dec_f32(unsigned u) {
    return (u & 0x80000000u) ? __uint_as_float(u ^ 0x80000000u) : __uint_as_float(~u);
}
__device__ __forceinline__ float leaky(float v) { return v > 0.0f ? v : NEG_SLOPE * v; }

__device__ __forceinline__ void atomic_add_f32(float* p, float v) {
    __hip_atomic_fetch_add(p, v, __ATOMIC_RELAXED, __HIP_MEMORY_SCOPE_AGENT);
}

// ---------------------------------------------------------------------------
// K1: h = x @ W_gat^T  (WMMA f16 in / f32 accum), fused a_s / a_d dot products
// block = 128 threads (4 waves), each block does a 64x128 output tile
// ---------------------------------------------------------------------------
__global__ __launch_bounds__(128)
void gat_proj_kernel(const float* __restrict__ x, const float* __restrict__ Wg,
                     const float* __restrict__ att_src, const float* __restrict__ att_dst,
                     float* __restrict__ h_out, float* __restrict__ a_s, float* __restrict__ a_d)
{
    __shared__ _Float16 Wl[128 * 128];   // W_gat, f16, row-major [j][k]  (32 KB)
    __shared__ _Float16 xl[ROWS_PER_BLOCK * 128];   // x tile, f16 (16 KB)
    __shared__ float    hl[ROWS_PER_BLOCK * 128];   // h tile, f32 (32 KB)

    const int t = threadIdx.x;
    const int row0 = blockIdx.x * ROWS_PER_BLOCK;

    for (int i = t; i < 128 * 128; i += 128) Wl[i] = (_Float16)Wg[i];
    for (int i = t; i < ROWS_PER_BLOCK * 128; i += 128) {
        int r = i >> 7, c = i & 127;
        int gr = row0 + r; if (gr >= N_NODES) gr = N_NODES - 1;   // clamp tail
        xl[i] = (_Float16)x[gr * 128 + c];
    }
    __syncthreads();

    const int wave = t >> 5;
    const int lane = t & 31;
    const int half = lane >> 4;
    const int l16  = lane & 15;
    const int mloc = wave * 16;

    for (int nt = 0; nt < 8; ++nt) {
        const int n0 = nt * 16;
        v8f c = {};
        #pragma unroll
        for (int kc = 0; kc < 128; kc += 32) {
            union { v16h v; unsigned u[8]; } A, B;
            #pragma unroll
            for (int vv = 0; vv < 8; ++vv) {
                // A: 16-bit 16x32 layout: VGPR v holds K = base + {0,1}
                int ka = kc + ((vv < 4) ? 0 : 16) + half * 8 + 2 * (vv & 3);
                A.u[vv] = *(const unsigned*)&xl[(mloc + l16) * 128 + ka];
                // B: same K order as A; B[k][n] = W[n][k]
                int kb = kc + ((vv < 4) ? 0 : 16) + half * 8 + 2 * (vv & 3);
                B.u[vv] = *(const unsigned*)&Wl[(n0 + l16) * 128 + kb];
            }
            c = __builtin_amdgcn_wmma_f32_16x16x32_f16(false, A.v, false, B.v,
                                                       (short)0, c, false, false);
        }
        #pragma unroll
        for (int r = 0; r < 8; ++r)
            hl[(mloc + half * 8 + r) * 128 + n0 + l16] = c[r];
    }
    __syncthreads();

    // store h tile (vectorized) and fused attention dot products
    for (int i = t; i < ROWS_PER_BLOCK * 32; i += 128) {
        int r = i >> 5, c4 = (i & 31) * 4;
        int gr = row0 + r;
        if (gr < N_NODES)
            *(float4*)&h_out[(size_t)gr * 128 + c4] = *(const float4*)&hl[r * 128 + c4];
    }
    for (int p = t; p < ROWS_PER_BLOCK * HEADS; p += 128) {
        int r = p >> 2, hh = p & 3;
        int gr = row0 + r;
        if (gr < N_NODES) {
            float ss = 0.0f, dd = 0.0f;
            #pragma unroll 8
            for (int dc = 0; dc < DDIM; ++dc) {
                float hv = hl[r * 128 + hh * DDIM + dc];
                ss += hv * att_src[hh * DDIM + dc];
                dd += hv * att_dst[hh * DDIM + dc];
            }
            a_s[gr * HEADS + hh] = ss;
            a_d[gr * HEADS + hh] = dd;
        }
    }
}

// ---------------------------------------------------------------------------
// K2: per-edge logits -> segment max via monotonic-uint atomicMax
// ---------------------------------------------------------------------------
__global__ __launch_bounds__(256)
void edge_max_kernel(const int* __restrict__ ei, const float* __restrict__ a_s,
                     const float* __restrict__ a_d, unsigned* __restrict__ m)
{
    const int Et = E_EDGES + N_NODES;
    int e = blockIdx.x * blockDim.x + threadIdx.x;
    if (e >= Et) return;
    int s, d;
    if (e < E_EDGES) { s = ei[e]; d = ei[E_EDGES + e]; } else { s = d = e - E_EDGES; }
    #pragma unroll
    for (int hh = 0; hh < HEADS; ++hh) {
        float l = leaky(a_s[s * HEADS + hh] + a_d[d * HEADS + hh]);
        atomicMax(&m[d * HEADS + hh], enc_f32(l));
    }
}

// ---------------------------------------------------------------------------
// K3: e = exp(logit - m[dst]); denom[dst] += e
// ---------------------------------------------------------------------------
__global__ __launch_bounds__(256)
void edge_denom_kernel(const int* __restrict__ ei, const float* __restrict__ a_s,
                       const float* __restrict__ a_d, const unsigned* __restrict__ m,
                       float* __restrict__ denom)
{
    const int Et = E_EDGES + N_NODES;
    int e = blockIdx.x * blockDim.x + threadIdx.x;
    if (e >= Et) return;
    int s, d;
    if (e < E_EDGES) { s = ei[e]; d = ei[E_EDGES + e]; } else { s = d = e - E_EDGES; }
    #pragma unroll
    for (int hh = 0; hh < HEADS; ++hh) {
        float l = leaky(a_s[s * HEADS + hh] + a_d[d * HEADS + hh]);
        float ex = __expf(l - dec_f32(m[d * HEADS + hh]));
        atomic_add_f32(&denom[d * HEADS + hh], ex);
    }
}

// ---------------------------------------------------------------------------
// K4: one wave per edge; lane i handles feature cols 4i..4i+3 (head = i>>3)
//     agg[dst] += alpha * h[src]
// ---------------------------------------------------------------------------
__global__ __launch_bounds__(256)
void edge_agg_kernel(const int* __restrict__ ei, const float* __restrict__ a_s,
                     const float* __restrict__ a_d, const unsigned* __restrict__ m,
                     const float* __restrict__ denom, const float* __restrict__ hfeat,
                     float* __restrict__ agg)
{
    const int Et = E_EDGES + N_NODES;
    const int lane = threadIdx.x & 31;
    const int wid  = (blockIdx.x * blockDim.x + threadIdx.x) >> 5;
    const int nw   = (gridDim.x * blockDim.x) >> 5;

    for (int e = wid; e < Et; e += nw) {
        int s, d;
        if (e < E_EDGES) { s = ei[e]; d = ei[E_EDGES + e]; } else { s = d = e - E_EDGES; }
        float w[HEADS];
        #pragma unroll
        for (int hh = 0; hh < HEADS; ++hh) {
            float l = leaky(a_s[s * HEADS + hh] + a_d[d * HEADS + hh]);
            w[hh] = __expf(l - dec_f32(m[d * HEADS + hh])) / denom[d * HEADS + hh];
        }
        const int c0 = lane * 4;
        const float wl = w[lane >> 3];
        float4 hv = *(const float4*)&hfeat[(size_t)s * 128 + c0];
        float* ap = &agg[(size_t)d * 128 + c0];
        atomic_add_f32(&ap[0], wl * hv.x);
        atomic_add_f32(&ap[1], wl * hv.y);
        atomic_add_f32(&ap[2], wl * hv.z);
        atomic_add_f32(&ap[3], wl * hv.w);
    }
}

// ---------------------------------------------------------------------------
// K5: out = relu(agg + bias_gat) @ W_lin^T + b_lin   (WMMA, relu fused in LDS stage)
// ---------------------------------------------------------------------------
__global__ __launch_bounds__(128)
void out_gemm_kernel(const float* __restrict__ agg, const float* __restrict__ bias_gat,
                     const float* __restrict__ Wlin, const float* __restrict__ blin,
                     float* __restrict__ out)
{
    __shared__ _Float16 Wl[128 * 128];
    __shared__ _Float16 rl[ROWS_PER_BLOCK * 128];
    __shared__ float    ol[ROWS_PER_BLOCK * 128];

    const int t = threadIdx.x;
    const int row0 = blockIdx.x * ROWS_PER_BLOCK;

    for (int i = t; i < 128 * 128; i += 128) Wl[i] = (_Float16)Wlin[i];
    for (int i = t; i < ROWS_PER_BLOCK * 128; i += 128) {
        int r = i >> 7, c = i & 127;
        int gr = row0 + r; if (gr >= N_NODES) gr = N_NODES - 1;
        float v = agg[(size_t)gr * 128 + c] + bias_gat[c];
        rl[i] = (_Float16)(v > 0.0f ? v : 0.0f);           // fused ReLU
    }
    __syncthreads();

    const int wave = t >> 5;
    const int lane = t & 31;
    const int half = lane >> 4;
    const int l16  = lane & 15;
    const int mloc = wave * 16;

    for (int nt = 0; nt < 8; ++nt) {
        const int n0 = nt * 16;
        v8f c = {};
        #pragma unroll
        for (int kc = 0; kc < 128; kc += 32) {
            union { v16h v; unsigned u[8]; } A, B;
            #pragma unroll
            for (int vv = 0; vv < 8; ++vv) {
                int ka = kc + ((vv < 4) ? 0 : 16) + half * 8 + 2 * (vv & 3);
                A.u[vv] = *(const unsigned*)&rl[(mloc + l16) * 128 + ka];
                int kb = kc + ((vv < 4) ? 0 : 16) + half * 8 + 2 * (vv & 3);
                B.u[vv] = *(const unsigned*)&Wl[(n0 + l16) * 128 + kb];
            }
            c = __builtin_amdgcn_wmma_f32_16x16x32_f16(false, A.v, false, B.v,
                                                       (short)0, c, false, false);
        }
        #pragma unroll
        for (int r = 0; r < 8; ++r)
            ol[(mloc + half * 8 + r) * 128 + n0 + l16] = c[r];
    }
    __syncthreads();

    for (int i = t; i < ROWS_PER_BLOCK * 32; i += 128) {
        int r = i >> 5, c4 = (i & 31) * 4;
        int gr = row0 + r;
        if (gr < N_NODES) {
            float4 o  = *(const float4*)&ol[r * 128 + c4];
            float4 bl = *(const float4*)&blin[c4];
            o.x += bl.x; o.y += bl.y; o.z += bl.z; o.w += bl.w;
            *(float4*)&out[(size_t)gr * 128 + c4] = o;
        }
    }
}

// ---------------------------------------------------------------------------
extern "C" void kernel_launch(void* const* d_in, const int* in_sizes, int n_in,
                              void* d_out, int out_size, void* d_ws, size_t ws_size,
                              hipStream_t stream) {
    const float* x        = (const float*)d_in[0];
    const int*   ei       = (const int*)  d_in[1];
    const float* Wg       = (const float*)d_in[2];
    const float* att_src  = (const float*)d_in[3];
    const float* att_dst  = (const float*)d_in[4];
    const float* bias_gat = (const float*)d_in[5];
    const float* Wlin     = (const float*)d_in[6];
    const float* blin     = (const float*)d_in[7];
    float* out = (float*)d_out;

    char* ws = (char*)d_ws;
    size_t off = 0;
    auto carve = [&](size_t bytes) -> void* {
        void* p = ws + off;
        off += (bytes + 255) & ~(size_t)255;
        return p;
    };
    float*    h     = (float*)   carve((size_t)N_NODES * 128 * sizeof(float));
    float*    a_s   = (float*)   carve((size_t)N_NODES * HEADS * sizeof(float));
    float*    a_d   = (float*)   carve((size_t)N_NODES * HEADS * sizeof(float));
    unsigned* m     = (unsigned*)carve((size_t)N_NODES * HEADS * sizeof(unsigned));
    float*    denom = (float*)   carve((size_t)N_NODES * HEADS * sizeof(float));
    float*    agg   = (float*)   carve((size_t)N_NODES * 128 * sizeof(float));

    // init: m=0 (monotonic encoding floor), denom=0, agg=0
    hipMemsetAsync(m,     0, (size_t)N_NODES * HEADS * sizeof(unsigned), stream);
    hipMemsetAsync(denom, 0, (size_t)N_NODES * HEADS * sizeof(float),    stream);
    hipMemsetAsync(agg,   0, (size_t)N_NODES * 128 * sizeof(float),      stream);

    gat_proj_kernel<<<N_BLOCKS, 128, 0, stream>>>(x, Wg, att_src, att_dst, h, a_s, a_d);

    const int Et = E_EDGES + N_NODES;
    edge_max_kernel  <<<(Et + 255) / 256, 256, 0, stream>>>(ei, a_s, a_d, m);
    edge_denom_kernel<<<(Et + 255) / 256, 256, 0, stream>>>(ei, a_s, a_d, m, denom);
    edge_agg_kernel  <<<8192, 256, 0, stream>>>(ei, a_s, a_d, m, denom, h, agg);

    out_gemm_kernel<<<N_BLOCKS, 128, 0, stream>>>(agg, bias_gat, Wlin, blin, out);
}
